// MolecularGNN_60911226192225
// MI455X (gfx1250) — matrix operand order repaired and checked by reading, not compiled, not run on hardware
//
#include <hip/hip_runtime.h>
#include <math.h>

// ---------------- problem constants (match reference) ----------------
#define NN      50000
#define EE      200000
#define GG      2000
#define F_IN    64
#define HID     128
#define HEADS   4
#define DM      512           // HID*HEADS
#define ET      (EE + NN)     // edges + self loops
#define NEG_SLOPE 0.2f
#define BN_EPS  1e-5f

typedef __attribute__((ext_vector_type(16))) _Float16 v16h;
typedef __attribute__((ext_vector_type(8)))  _Float16 v8h;
typedef __attribute__((ext_vector_type(4)))  _Float16 h4;
typedef __attribute__((ext_vector_type(8)))  float    v8f;

#define CDIV(a,b) (((a)+(b)-1)/(b))
#define LDS_STRIDE 40   // halfs per row: 80 B -> 16B-aligned fragments, bank-spread

// ---------------- helpers ----------------
__device__ __forceinline__ void atomicMaxF(float* addr, float val) {
    // works for any mix of signs when slot initialized to -inf
    if (val >= 0.0f) atomicMax((int*)addr, __float_as_int(val));
    else             atomicMin((unsigned int*)addr, __float_as_uint(val));
}

__device__ __forceinline__ v16h pack16(v8h lo, v8h hi) {
    v16h r;
#pragma unroll
    for (int i = 0; i < 8; ++i) { r[i] = lo[i]; r[8 + i] = hi[i]; }
    return r;
}

__global__ void fill_kernel(float* p, float v, size_t n) {
    size_t i = (size_t)blockIdx.x * blockDim.x + threadIdx.x;
    if (i < n) p[i] = v;
}

// ---------------- WMMA GEMM: C[M,cols] = A[M,K] @ B[K,cols] ----------------
// f32 in, f16 MACs, f32 accumulate. 256 threads / 8 waves per block.
// Block tile: 128 rows x 128 cols. Wave tile: 32 rows x 64 cols = 2x4 WMMA accs.
// A(128x32) and B(32x128) staged per K-step through LDS as f16 (padded rows).
// K multiple of 32. Partial row/col blocks handled by clamp (loads) + guard (stores).
__global__ void __launch_bounds__(256)
gemm_wmma_tiled(const float* __restrict__ A, const float* __restrict__ B,
                float* __restrict__ C, int M, int K, int cols) {
    __shared__ __align__(16) _Float16 At[128 * LDS_STRIDE];   // [m][k]
    __shared__ __align__(16) _Float16 Bt[128 * LDS_STRIDE];   // [n][k] (transposed)

    const int tid   = threadIdx.x;
    const int lane  = tid & 31;
    const int wave  = tid >> 5;
    const int waveM = wave >> 1;            // 0..3 -> 32-row band
    const int waveN = wave & 1;             // 0..1 -> 64-col band
    const int rowBase  = blockIdx.x * 128;
    const int colBlock = blockIdx.y * 128;

    const int half = (lane < 16) ? 0 : 1;   // lane half selects K sub-block
    const int l15  = lane & 15;
    const int kbA  = half * 8;              // A frag: k 0-7/8-15 (+16)
    const int kbB  = half * 8;              // B frag: k 0-7/8-15 (+16)

    v8f acc[2][4] = {};

    for (int k0 = 0; k0 < K; k0 += 32) {
        // ---- stage A: 128 rows x 32 k  (1024 float4, coalesced) ----
        for (int e = tid; e < 1024; e += 256) {
            int m = e >> 3;                 // 8 float4 per row
            int k = (e & 7) * 4;
            int row = rowBase + m; if (row >= M) row = M - 1;   // clamp (safe re-read)
            float4 v = *(const float4*)(A + (size_t)row * K + k0 + k);
            h4 hv; hv[0] = (_Float16)v.x; hv[1] = (_Float16)v.y;
            hv[2] = (_Float16)v.z; hv[3] = (_Float16)v.w;
            *(h4*)(&At[m * LDS_STRIDE + k]) = hv;
        }
        // ---- stage B: 32 k x 128 n, transposed into Bt[n][k] ----
        if (colBlock + 128 <= cols) {       // fast path: coalesced float4 on n
            for (int e = tid; e < 1024; e += 256) {
                int k = e >> 5;             // 32 float4 per k-row
                int n = (e & 31) * 4;
                float4 v = *(const float4*)(B + (size_t)(k0 + k) * cols + colBlock + n);
                Bt[(n + 0) * LDS_STRIDE + k] = (_Float16)v.x;
                Bt[(n + 1) * LDS_STRIDE + k] = (_Float16)v.y;
                Bt[(n + 2) * LDS_STRIDE + k] = (_Float16)v.z;
                Bt[(n + 3) * LDS_STRIDE + k] = (_Float16)v.w;
            }
        } else {                            // narrow matrix: scalar with clamp
            for (int e = tid; e < 32 * 128; e += 256) {
                int k = e >> 7, n = e & 127;
                int col = colBlock + n; if (col >= cols) col = cols - 1;
                Bt[n * LDS_STRIDE + k] = (_Float16)B[(size_t)(k0 + k) * cols + col];
            }
        }
        __syncthreads();

        // ---- fragments + 8 WMMAs ----
        v16h afr[2];
#pragma unroll
        for (int t = 0; t < 2; ++t) {
            int m = waveM * 32 + t * 16 + l15;
            v8h lo = *(const v8h*)(&At[m * LDS_STRIDE + kbA]);
            v8h hi = *(const v8h*)(&At[m * LDS_STRIDE + 16 + kbA]);
            afr[t] = pack16(lo, hi);
        }
#pragma unroll
        for (int u = 0; u < 4; ++u) {
            int n = waveN * 64 + u * 16 + l15;
            v8h blo = *(const v8h*)(&Bt[n * LDS_STRIDE + kbB]);
            v8h bhi = *(const v8h*)(&Bt[n * LDS_STRIDE + 16 + kbB]);
            v16h bfr = pack16(blo, bhi);
            acc[0][u] = __builtin_amdgcn_wmma_f32_16x16x32_f16(false, afr[0], false, bfr,
                                                               (short)0, acc[0][u], false, false);
            acc[1][u] = __builtin_amdgcn_wmma_f32_16x16x32_f16(false, afr[1], false, bfr,
                                                               (short)0, acc[1][u], false, false);
        }
        __syncthreads();
    }

    // ---- store D tiles (C/D layout: VGPR r -> row r + 8*half) ----
#pragma unroll
    for (int t = 0; t < 2; ++t) {
#pragma unroll
        for (int u = 0; u < 4; ++u) {
            int col = colBlock + waveN * 64 + u * 16 + l15;
            if (col >= cols) continue;
#pragma unroll
            for (int r = 0; r < 8; ++r) {
                int row = rowBase + waveM * 32 + t * 16 + half * 8 + r;
                if (row < M) C[(size_t)row * cols + col] = acc[t][u][r];
            }
        }
    }
}

// ---------------- attention scalar scores: s_src/s_dst [N,HEADS] ----------------
__global__ void att_scores(const float* __restrict__ h, const float* __restrict__ a_s,
                           const float* __restrict__ a_d, float* __restrict__ ssrc,
                           float* __restrict__ sdst) {
    int idx = blockIdx.x * blockDim.x + threadIdx.x;     // node*HEADS + head
    if (idx >= NN * HEADS) return;
    int node = idx >> 2, hd = idx & 3;
    const float* hp = h + (size_t)node * DM + hd * HID;
    const float* as = a_s + hd * HID;
    const float* ad = a_d + hd * HID;
    float s1 = 0.f, s2 = 0.f;
    for (int d = 0; d < HID; ++d) { float v = hp[d]; s1 += v * as[d]; s2 += v * ad[d]; }
    ssrc[idx] = s1; sdst[idx] = s2;
}

__device__ __forceinline__ void edge_ends(const int* ei, int t, int& s, int& d) {
    if (t < EE) { s = ei[t]; d = ei[EE + t]; }
    else        { s = t - EE; d = t - EE; }          // self loop
}

// leaky-relu score + segment max over dst
__global__ void edge_score_max(const int* __restrict__ ei, const float* __restrict__ ssrc,
                               const float* __restrict__ sdst, float* __restrict__ e_out,
                               float* __restrict__ m) {
    int t = blockIdx.x * blockDim.x + threadIdx.x;
    if (t >= ET) return;
    int s, d; edge_ends(ei, t, s, d);
#pragma unroll
    for (int hd = 0; hd < HEADS; ++hd) {
        float e = ssrc[s * HEADS + hd] + sdst[d * HEADS + hd];
        e = (e > 0.f) ? e : NEG_SLOPE * e;
        e_out[(size_t)t * HEADS + hd] = e;
        atomicMaxF(&m[d * HEADS + hd], e);
    }
}

// exp(e - max) + segment sum over dst
__global__ void edge_exp_sum(const int* __restrict__ ei, float* __restrict__ e_out,
                             const float* __restrict__ m, float* __restrict__ denom) {
    int t = blockIdx.x * blockDim.x + threadIdx.x;
    if (t >= ET) return;
    int s, d; edge_ends(ei, t, s, d);
#pragma unroll
    for (int hd = 0; hd < HEADS; ++hd) {
        float ex = __expf(e_out[(size_t)t * HEADS + hd] - m[d * HEADS + hd]);
        e_out[(size_t)t * HEADS + hd] = ex;
        atomicAdd(&denom[d * HEADS + hd], ex);
    }
}

__global__ void edge_norm(const int* __restrict__ ei, float* __restrict__ e_out,
                          const float* __restrict__ denom) {
    int t = blockIdx.x * blockDim.x + threadIdx.x;
    if (t >= ET) return;
    int s, d; edge_ends(ei, t, s, d);
#pragma unroll
    for (int hd = 0; hd < HEADS; ++hd)
        e_out[(size_t)t * HEADS + hd] /= denom[d * HEADS + hd];
}

// out[dst] += alpha * h[src]  (thread = (edge, 4 features)) — the bandwidth hot loop
__global__ void aggregate(const int* __restrict__ ei, const float* __restrict__ h,
                          const float* __restrict__ alpha, float* __restrict__ out) {
    size_t idx = (size_t)blockIdx.x * blockDim.x + threadIdx.x;  // t*128 + q
    if (idx >= (size_t)ET * (DM / 4)) return;
    int t = (int)(idx >> 7);
    int q = (int)(idx & 127);
    int s, d; edge_ends(ei, t, s, d);
    int c  = q * 4;
    int hd = c >> 7;                       // which head this feature chunk is in
    float w = alpha[(size_t)t * HEADS + hd];
    float4 hv = *(const float4*)(h + (size_t)s * DM + c);
    float* op = out + (size_t)d * DM + c;
    atomicAdd(op + 0, hv.x * w);
    atomicAdd(op + 1, hv.y * w);
    atomicAdd(op + 2, hv.z * w);
    atomicAdd(op + 3, hv.w * w);
}

// ---------------- batch norm (biased var, like jnp.var) ----------------
__global__ void bn_stats(const float* __restrict__ x, float* __restrict__ mean,
                         float* __restrict__ rstd) {
    int f = blockIdx.x;                        // feature 0..511
    float s = 0.f, s2 = 0.f;
    for (int i = threadIdx.x; i < NN; i += blockDim.x) {
        float v = x[(size_t)i * DM + f];
        s += v; s2 += v * v;
    }
    __shared__ float sh[256], sh2[256];
    sh[threadIdx.x] = s; sh2[threadIdx.x] = s2;
    __syncthreads();
    for (int o = blockDim.x >> 1; o > 0; o >>= 1) {
        if (threadIdx.x < o) { sh[threadIdx.x] += sh[threadIdx.x + o];
                               sh2[threadIdx.x] += sh2[threadIdx.x + o]; }
        __syncthreads();
    }
    if (threadIdx.x == 0) {
        float mu  = sh[0] / (float)NN;
        float var = sh2[0] / (float)NN - mu * mu;
        mean[f] = mu;
        rstd[f] = rsqrtf(var + BN_EPS);
    }
}

__global__ void bn_relu(float* __restrict__ x, const float* __restrict__ mean,
                        const float* __restrict__ rstd, const float* __restrict__ g,
                        const float* __restrict__ be) {
    size_t i = (size_t)blockIdx.x * blockDim.x + threadIdx.x;
    if (i >= (size_t)NN * DM) return;
    int f = (int)(i & (DM - 1));
    float y = (x[i] - mean[f]) * rstd[f] * g[f] + be[f];
    x[i] = fmaxf(y, 0.f);
}

// ---------------- graph pooling into [G, 2*DM] concat(mean, max) ----------------
__global__ void pool_init(float* __restrict__ pooled) {
    size_t i = (size_t)blockIdx.x * blockDim.x + threadIdx.x;
    if (i >= (size_t)GG * (2 * DM)) return;
    pooled[i] = ((i & (2 * DM - 1)) < DM) ? 0.f : -__builtin_inff();
}

__global__ void pool_count(const int* __restrict__ batch, float* __restrict__ cnt) {
    int n = blockIdx.x * blockDim.x + threadIdx.x;
    if (n >= NN) return;
    atomicAdd(&cnt[batch[n]], 1.0f);
}

__global__ void pool_scatter(const float* __restrict__ x, const int* __restrict__ batch,
                             float* __restrict__ pooled) {
    size_t i = (size_t)blockIdx.x * blockDim.x + threadIdx.x;
    if (i >= (size_t)NN * DM) return;
    int n = (int)(i >> 9);
    int c = (int)(i & (DM - 1));
    int b = batch[n];
    float v = x[i];
    atomicAdd(&pooled[(size_t)b * (2 * DM) + c], v);
    atomicMaxF(&pooled[(size_t)b * (2 * DM) + DM + c], v);
}

__global__ void pool_finalize(float* __restrict__ pooled, const float* __restrict__ cnt) {
    size_t i = (size_t)blockIdx.x * blockDim.x + threadIdx.x;
    if (i >= (size_t)GG * DM) return;
    int g = (int)(i >> 9);
    int c = (int)(i & (DM - 1));
    float ct = cnt[g];
    pooled[(size_t)g * (2 * DM) + c] /= fmaxf(ct, 1.f);
    float* mp = &pooled[(size_t)g * (2 * DM) + DM + c];
    if (ct <= 0.f) *mp = 0.f;
}

// ---------------- classifier tail ----------------
__global__ void bias_relu(float* __restrict__ x, const float* __restrict__ b,
                          int cols, size_t n) {
    size_t i = (size_t)blockIdx.x * blockDim.x + threadIdx.x;
    if (i >= n) return;
    int c = (int)(i % cols);
    x[i] = fmaxf(x[i] + b[c], 0.f);
}

__global__ void final_mlp(const float* __restrict__ h2, const float* __restrict__ w,
                          const float* __restrict__ b, float* __restrict__ out) {
    int g = blockIdx.x * blockDim.x + threadIdx.x;
    if (g >= GG) return;
    float s = b[0];
    for (int d = 0; d < 64; ++d) s += h2[g * 64 + d] * w[d];
    out[g] = 1.f / (1.f + __expf(-s));
}

// ---------------- host-side orchestration ----------------
extern "C" void kernel_launch(void* const* d_in, const int* in_sizes, int n_in,
                              void* d_out, int out_size, void* d_ws, size_t ws_size,
                              hipStream_t stream) {
    const float* x     = (const float*)d_in[0];
    const int*   ei    = (const int*)d_in[1];
    const int*   batch = (const int*)d_in[2];
    const float* W[3]  = {(const float*)d_in[3],  (const float*)d_in[9],  (const float*)d_in[15]};
    const float* AS[3] = {(const float*)d_in[4],  (const float*)d_in[10], (const float*)d_in[16]};
    const float* AD[3] = {(const float*)d_in[5],  (const float*)d_in[11], (const float*)d_in[17]};
    // b_l (d_in[6,12,18]) cancels under BatchNorm -> skipped
    const float* Gm[3] = {(const float*)d_in[7],  (const float*)d_in[13], (const float*)d_in[19]};
    const float* Be[3] = {(const float*)d_in[8],  (const float*)d_in[14], (const float*)d_in[20]};
    const float* c1w = (const float*)d_in[21]; const float* c1b = (const float*)d_in[22];
    const float* c2w = (const float*)d_in[23]; const float* c2b = (const float*)d_in[24];
    const float* c3w = (const float*)d_in[25]; const float* c3b = (const float*)d_in[26];
    float* out = (float*)d_out;

    // workspace carve-up (floats)
    float* ws = (float*)d_ws;
    size_t o = 0;
    float* bufH   = ws + o; o += (size_t)NN * DM;       // GEMM output / attention input
    float* bufX   = ws + o; o += (size_t)NN * DM;       // aggregation output / next input
    float* ssrc   = ws + o; o += (size_t)NN * HEADS;
    float* sdst   = ws + o; o += (size_t)NN * HEADS;
    float* segmax = ws + o; o += (size_t)NN * HEADS;
    float* denom  = ws + o; o += (size_t)NN * HEADS;
    float* alpha  = ws + o; o += (size_t)ET * HEADS;
    float* mean   = ws + o; o += DM;
    float* rstd   = ws + o; o += DM;
    float* pooled = ws + o; o += (size_t)GG * (2 * DM);
    float* cnt    = ws + o; o += GG;
    float* h1     = ws + o; o += (size_t)GG * 128;
    float* h2     = ws + o; o += (size_t)GG * 64;

    const int TB = 256;
    const float* cur = x;
    int K = F_IN;

    for (int l = 0; l < 3; ++l) {
        // h = cur @ W_l   (WMMA, 128x128 block tiles)
        dim3 gg(CDIV(NN, 128), CDIV(DM, 128));
        gemm_wmma_tiled<<<gg, TB, 0, stream>>>(cur, W[l], bufH, NN, K, DM);

        // attention scalars
        att_scores<<<CDIV(NN * HEADS, TB), TB, 0, stream>>>(bufH, AS[l], AD[l], ssrc, sdst);

        // init segment buffers + aggregation target
        fill_kernel<<<CDIV(NN * HEADS, TB), TB, 0, stream>>>(segmax, -__builtin_inff(), (size_t)NN * HEADS);
        fill_kernel<<<CDIV(NN * HEADS, TB), TB, 0, stream>>>(denom, 0.f, (size_t)NN * HEADS);
        fill_kernel<<<CDIV((size_t)NN * DM, TB), TB, 0, stream>>>(bufX, 0.f, (size_t)NN * DM);

        // segment softmax over incoming edges of each dst
        edge_score_max<<<CDIV(ET, TB), TB, 0, stream>>>(ei, ssrc, sdst, alpha, segmax);
        edge_exp_sum  <<<CDIV(ET, TB), TB, 0, stream>>>(ei, alpha, segmax, denom);
        edge_norm     <<<CDIV(ET, TB), TB, 0, stream>>>(ei, alpha, denom);

        // weighted scatter-add of h[src] into bufX[dst]
        aggregate<<<CDIV((size_t)ET * (DM / 4), TB), TB, 0, stream>>>(ei, bufH, alpha, bufX);

        // batch norm + relu (in place)
        bn_stats<<<DM, TB, 0, stream>>>(bufX, mean, rstd);
        bn_relu <<<CDIV((size_t)NN * DM, TB), TB, 0, stream>>>(bufX, mean, rstd, Gm[l], Be[l]);

        cur = bufX;
        K = DM;
    }

    // graph pooling -> pooled[G, 1024] = concat(mean, max)
    pool_init    <<<CDIV((size_t)GG * 2 * DM, TB), TB, 0, stream>>>(pooled);
    fill_kernel  <<<CDIV(GG, TB), TB, 0, stream>>>(cnt, 0.f, (size_t)GG);
    pool_count   <<<CDIV(NN, TB), TB, 0, stream>>>(batch, cnt);
    pool_scatter <<<CDIV((size_t)NN * DM, TB), TB, 0, stream>>>(bufX, batch, pooled);
    pool_finalize<<<CDIV((size_t)GG * DM, TB), TB, 0, stream>>>(pooled, cnt);

    // classifier MLP: first two layers via WMMA GEMM
    {
        dim3 g1(CDIV(GG, 128), CDIV(128, 128));
        gemm_wmma_tiled<<<g1, TB, 0, stream>>>(pooled, c1w, h1, GG, 2 * DM, 128);
        bias_relu<<<CDIV((size_t)GG * 128, TB), TB, 0, stream>>>(h1, c1b, 128, (size_t)GG * 128);

        dim3 g2(CDIV(GG, 128), CDIV(64, 128));
        gemm_wmma_tiled<<<g2, TB, 0, stream>>>(h1, c2w, h2, GG, 128, 64);
        bias_relu<<<CDIV((size_t)GG * 64, TB), TB, 0, stream>>>(h2, c2b, 64, (size_t)GG * 64);

        final_mlp<<<CDIV(GG, TB), TB, 0, stream>>>(h2, c3w, c3b, out);
    }
}
